// SlotAttention_27504970563676
// MI455X (gfx1250) — matrix operand order repaired and checked by reading, not compiled, not run on hardware
//
#include <hip/hip_runtime.h>
#include <math.h>

#define Bb 64
#define Nn 4096
#define Dd 256
#define Ss 11
#define Kk 128
#define Hh 256
#define LN_EPS 1e-5f
#define EPSC 1e-6f

typedef __attribute__((ext_vector_type(16))) __bf16 v16bf;
typedef __attribute__((ext_vector_type(8)))  __bf16 v8bf;
typedef __attribute__((ext_vector_type(4)))  __bf16 v4bf;
typedef __attribute__((ext_vector_type(8)))  float  v8f;

union ABf { v16bf v; v8bf h[2]; };

// ---------------- K0: convert Wk/Wv to bf16 ----------------
__global__ void conv_w(const float* __restrict__ wk, const float* __restrict__ wv,
                       __bf16* __restrict__ wkb, __bf16* __restrict__ wvb) {
    int i = blockIdx.x * 256 + threadIdx.x;
    if (i < Kk * Dd) { wkb[i] = (__bf16)wk[i]; wvb[i] = (__bf16)wv[i]; }
}

// ---------------- K1: slots = mu + exp(log_sigma) * init ----------------
__global__ void init_slots(const float* __restrict__ si, const float* __restrict__ mu,
                           const float* __restrict__ ls, float* __restrict__ slots) {
    int i = blockIdx.x * 256 + threadIdx.x;
    if (i < Bb * Ss * Kk) {
        int k = i % Kk;
        slots[i] = mu[k] + __expf(ls[k]) * si[i];
    }
}

// ---------------- K2: fused LayerNorm + k/v projection (WMMA bf16) ----------------
// grid = B*N/128 blocks, 256 threads (8 waves); wave owns 16 rows.
__global__ void __launch_bounds__(256) ln_proj_kv(
    const float* __restrict__ x, const float* __restrict__ lnw, const float* __restrict__ lnb,
    const __bf16* __restrict__ wk, const __bf16* __restrict__ wv,
    __bf16* __restrict__ kout, __bf16* __restrict__ vout)
{
    __shared__ __bf16 tile[8][16][Dd];          // 64KB: wave-private 16x256 bf16 tiles
    const int lane = threadIdx.x & 31;
    const int wave = threadIdx.x >> 5;
    const int hf   = lane >> 4;                 // lane half (0/1)
    const int mr   = lane & 15;
    const size_t row0 = (size_t)blockIdx.x * 128 + wave * 16;
    const float* xw = x + row0 * Dd;
    const int c0 = lane * 8;
    const float4 w0 = *(const float4*)(lnw + c0);
    const float4 w1 = *(const float4*)(lnw + c0 + 4);
    const float4 b0 = *(const float4*)(lnb + c0);
    const float4 b1 = *(const float4*)(lnb + c0 + 4);

    for (int r = 0; r < 16; ++r) {
        const float4 v0 = *(const float4*)(xw + r * Dd + c0);
        const float4 v1 = *(const float4*)(xw + r * Dd + c0 + 4);
        float s  = v0.x + v0.y + v0.z + v0.w + v1.x + v1.y + v1.z + v1.w;
        float s2 = v0.x*v0.x + v0.y*v0.y + v0.z*v0.z + v0.w*v0.w
                 + v1.x*v1.x + v1.y*v1.y + v1.z*v1.z + v1.w*v1.w;
        #pragma unroll
        for (int off = 16; off; off >>= 1) {
            s  += __shfl_xor(s,  off, 32);
            s2 += __shfl_xor(s2, off, 32);
        }
        const float mean = s * (1.0f / Dd);
        const float rstd = rsqrtf(s2 * (1.0f / Dd) - mean * mean + LN_EPS);
        v8bf o;
        o[0] = (__bf16)((v0.x - mean) * rstd * w0.x + b0.x);
        o[1] = (__bf16)((v0.y - mean) * rstd * w0.y + b0.y);
        o[2] = (__bf16)((v0.z - mean) * rstd * w0.z + b0.z);
        o[3] = (__bf16)((v0.w - mean) * rstd * w0.w + b0.w);
        o[4] = (__bf16)((v1.x - mean) * rstd * w1.x + b1.x);
        o[5] = (__bf16)((v1.y - mean) * rstd * w1.y + b1.y);
        o[6] = (__bf16)((v1.z - mean) * rstd * w1.z + b1.z);
        o[7] = (__bf16)((v1.w - mean) * rstd * w1.w + b1.w);
        *(v8bf*)&tile[wave][r][c0] = o;
    }
    __syncthreads();

    // A operand as loaded here (contiguous k order, same as B): lane M=mr; elems 0..15 -> K=hf*16+e
    const __bf16* tA0 = &tile[wave][mr][hf * 16];
    const __bf16* tA1 = &tile[wave][mr][hf * 16 + 8];
    for (int ct = 0; ct < 8; ++ct) {
        v8f ak = {0.f,0.f,0.f,0.f,0.f,0.f,0.f,0.f};
        v8f av = {0.f,0.f,0.f,0.f,0.f,0.f,0.f,0.f};
        const int col = ct * 16 + mr;           // output feature (B-matrix N = mr)
        const __bf16* wkp = wk + (size_t)col * Dd + hf * 16;  // B elems: K = hf*16 + e
        const __bf16* wvp = wv + (size_t)col * Dd + hf * 16;
        #pragma unroll
        for (int ks = 0; ks < 8; ++ks) {        // K-dim 256 = 8 x 32
            ABf a;
            a.h[0] = *(const v8bf*)(tA0 + ks * 32);
            a.h[1] = *(const v8bf*)(tA1 + ks * 32);
            const v16bf bk = *(const v16bf*)(wkp + ks * 32);
            const v16bf bv = *(const v16bf*)(wvp + ks * 32);
            ak = __builtin_amdgcn_wmma_f32_16x16x32_bf16(false, a.v, false, bk, (short)0, ak, false, false);
            av = __builtin_amdgcn_wmma_f32_16x16x32_bf16(false, a.v, false, bv, (short)0, av, false, false);
        }
        #pragma unroll
        for (int r = 0; r < 8; ++r) {           // C/D: M = r + 8*hf, N = mr
            const size_t n = row0 + r + 8 * hf;
            kout[n * Kk + col] = (__bf16)ak[r];
            vout[n * Kk + col] = (__bf16)av[r];
        }
    }
}

// ---------------- K3: per-batch q = LN(slots)*Wq^T, zero accumulators ----------------
__global__ void __launch_bounds__(128) qgen_zero(
    const float* __restrict__ slots, const float* __restrict__ lsw, const float* __restrict__ lsb,
    const float* __restrict__ Wq, __bf16* __restrict__ qbf,
    float* __restrict__ accum, float* __restrict__ denom)
{
    __shared__ float sn[Ss][Kk];
    const int b = blockIdx.x, t = threadIdx.x;
    const int lane = t & 31, wave = t >> 5;
    float* ac = accum + (size_t)b * Ss * Kk;
    for (int i = t; i < Ss * Kk; i += 128) ac[i] = 0.f;
    if (t < Ss) denom[b * Ss + t] = 0.f;
    const float* sl = slots + (size_t)b * Ss * Kk;
    for (int s = wave; s < Ss; s += 4) {
        const int c = lane * 4;
        const float4 v0 = *(const float4*)(sl + s * Kk + c);
        float su = v0.x + v0.y + v0.z + v0.w;
        float s2 = v0.x*v0.x + v0.y*v0.y + v0.z*v0.z + v0.w*v0.w;
        #pragma unroll
        for (int off = 16; off; off >>= 1) {
            su += __shfl_xor(su, off, 32);
            s2 += __shfl_xor(s2, off, 32);
        }
        const float mean = su * (1.f / Kk);
        const float rstd = rsqrtf(s2 * (1.f / Kk) - mean * mean + LN_EPS);
        sn[s][c + 0] = (v0.x - mean) * rstd * lsw[c + 0] + lsb[c + 0];
        sn[s][c + 1] = (v0.y - mean) * rstd * lsw[c + 1] + lsb[c + 1];
        sn[s][c + 2] = (v0.z - mean) * rstd * lsw[c + 2] + lsb[c + 2];
        sn[s][c + 3] = (v0.w - mean) * rstd * lsw[c + 3] + lsb[c + 3];
    }
    __syncthreads();
    const int j = t;
    const float* wr = Wq + (size_t)j * Kk;
    for (int s = 0; s < Ss; ++s) {
        float acc = 0.f;
        for (int kk = 0; kk < Kk; ++kk) acc += sn[s][kk] * wr[kk];
        qbf[((size_t)b * 16 + s) * Kk + j] = (__bf16)acc;
    }
    for (int s = Ss; s < 16; ++s) qbf[((size_t)b * 16 + s) * Kk + j] = (__bf16)0.f;
}

// ---------------- K4: logits (WMMA) + softmax + renorm partials + attn^T*v ----------------
// grid = B*4 blocks, 256 threads (8 waves); wave owns 128 n rows (4 chunks of 32).
__global__ void __launch_bounds__(256) attn_iter(
    const __bf16* __restrict__ kbf, const __bf16* __restrict__ vbf,
    const __bf16* __restrict__ qbf,
    float* __restrict__ accum, float* __restrict__ denom)
{
    __shared__ float ptile[8][32][16];
    const int lane = threadIdx.x & 31;
    const int wave = threadIdx.x >> 5;
    const int hf = lane >> 4;
    const int sc = lane & 15;                 // s-column of logits tile (also A-row id)
    const int b = blockIdx.x >> 2;
    const int part = blockIdx.x & 3;
    const size_t base = (size_t)b * Nn;
    const int nw0 = part * 1024 + wave * 128;
    const float scale = 0.08838834764831845f; // 128^-0.5
    const __bf16* qrow = qbf + ((size_t)b * 16 + sc) * Kk + hf * 16;
    float uacc[44];
    #pragma unroll
    for (int i = 0; i < 44; ++i) uacc[i] = 0.f;
    float dacc = 0.f;
    const int k4 = lane * 4;
    float (*pw)[16] = ptile[wave];

    for (int ch = 0; ch < 4; ++ch) {
        const int nch = nw0 + ch * 32;
        #pragma unroll
        for (int sub = 0; sub < 2; ++sub) {
            const int n0 = nch + sub * 16;
            const __bf16* ap0 = kbf + (base + n0 + sc) * Kk + hf * 16;
            v8f acc = {0.f,0.f,0.f,0.f,0.f,0.f,0.f,0.f};
            #pragma unroll
            for (int ks = 0; ks < 4; ++ks) {   // K-dim 128 = 4 x 32
                ABf a;
                a.h[0] = *(const v8bf*)(ap0 + ks * 32);
                a.h[1] = *(const v8bf*)(ap0 + 8 + ks * 32);
                const v16bf bq = *(const v16bf*)(qrow + ks * 32);
                acc = __builtin_amdgcn_wmma_f32_16x16x32_bf16(false, a.v, false, bq, (short)0, acc, false, false);
            }
            #pragma unroll
            for (int r = 0; r < 8; ++r) {      // row n = n0 + r + 8*hf, col s = sc
                float lg = (sc < Ss) ? acc[r] * scale : -1e30f;
                float mx = lg;
                #pragma unroll
                for (int off = 1; off <= 8; off <<= 1) mx = fmaxf(mx, __shfl_xor(mx, off, 32));
                float e = __expf(lg - mx);
                float sm = e;
                #pragma unroll
                for (int off = 1; off <= 8; off <<= 1) sm += __shfl_xor(sm, off, 32);
                const float p = (sc < Ss) ? (e / sm + EPSC) : 0.f;
                dacc += p;
                pw[sub * 16 + r + 8 * hf][sc] = p;
            }
        }
        // updates partial: uacc[s][j] += p[n][s] * v[n][k4+j]
        const __bf16* vrow = vbf + (base + nch) * Kk + k4;
        for (int nn = 0; nn < 32; ++nn) {
            const v4bf vv = *(const v4bf*)(vrow + (size_t)nn * Kk);
            const float f0 = (float)vv[0], f1 = (float)vv[1];
            const float f2 = (float)vv[2], f3 = (float)vv[3];
            #pragma unroll
            for (int s = 0; s < Ss; ++s) {
                const float ps = pw[nn][s];
                uacc[s * 4 + 0] += ps * f0;
                uacc[s * 4 + 1] += ps * f1;
                uacc[s * 4 + 2] += ps * f2;
                uacc[s * 4 + 3] += ps * f3;
            }
        }
    }
    dacc += __shfl_xor(dacc, 16, 32);
    if (hf == 0 && sc < Ss) atomicAdd(&denom[b * Ss + sc], dacc);
    float* ab = accum + (size_t)b * Ss * Kk;
    #pragma unroll
    for (int s = 0; s < Ss; ++s) {
        #pragma unroll
        for (int j = 0; j < 4; ++j) atomicAdd(&ab[s * Kk + k4 + j], uacc[s * 4 + j]);
    }
}

// ---------------- K5: updates finalize + GRU + LN + MLP ----------------
__global__ void __launch_bounds__(256) gru_mlp(
    const float* __restrict__ accum, const float* __restrict__ denom,
    float* __restrict__ slots,
    const float* __restrict__ w_ih, const float* __restrict__ w_hh,
    const float* __restrict__ b_ih, const float* __restrict__ b_hh,
    const float* __restrict__ lmw, const float* __restrict__ lmb,
    const float* __restrict__ w1, const float* __restrict__ bb1,
    const float* __restrict__ w2, const float* __restrict__ bb2,
    float* __restrict__ out)
{
    __shared__ float upd[Ss][Kk], prev[Ss][Kk], gru[Ss][Kk], mn[Ss][Kk], hid[Ss][Hh];
    const int b = blockIdx.x, t = threadIdx.x;
    const int lane = t & 31, wave = t >> 5;
    const float* ac = accum + (size_t)b * Ss * Kk;
    const float* sl = slots + (size_t)b * Ss * Kk;
    for (int i = t; i < Ss * Kk; i += 256) {
        const int s = i / Kk;
        upd[s][i % Kk] = ac[i] / denom[b * Ss + s];
        prev[s][i % Kk] = sl[i];
    }
    __syncthreads();
    for (int i = t; i < Ss * Kk; i += 256) {
        const int s = i / Kk, k = i % Kk;
        float gir = b_ih[k], giz = b_ih[Kk + k], gin = b_ih[2 * Kk + k];
        float ghr = b_hh[k], ghz = b_hh[Kk + k], ghn = b_hh[2 * Kk + k];
        const float* wir = w_ih + (size_t)k * Kk;
        const float* wiz = w_ih + (size_t)(Kk + k) * Kk;
        const float* win = w_ih + (size_t)(2 * Kk + k) * Kk;
        const float* whr = w_hh + (size_t)k * Kk;
        const float* whz = w_hh + (size_t)(Kk + k) * Kk;
        const float* whn = w_hh + (size_t)(2 * Kk + k) * Kk;
        for (int kk = 0; kk < Kk; ++kk) {
            const float u = upd[s][kk], p = prev[s][kk];
            gir += u * wir[kk]; giz += u * wiz[kk]; gin += u * win[kk];
            ghr += p * whr[kk]; ghz += p * whz[kk]; ghn += p * whn[kk];
        }
        const float r = 1.f / (1.f + __expf(-(gir + ghr)));
        const float z = 1.f / (1.f + __expf(-(giz + ghz)));
        const float ng = tanhf(gin + r * ghn);
        gru[s][k] = (1.f - z) * ng + z * prev[s][k];
    }
    __syncthreads();
    for (int s = wave; s < Ss; s += 8) {
        const int c = lane * 4;
        const float4 v0 = *(const float4*)(&gru[s][c]);
        float su = v0.x + v0.y + v0.z + v0.w;
        float s2 = v0.x*v0.x + v0.y*v0.y + v0.z*v0.z + v0.w*v0.w;
        #pragma unroll
        for (int off = 16; off; off >>= 1) {
            su += __shfl_xor(su, off, 32);
            s2 += __shfl_xor(s2, off, 32);
        }
        const float mean = su * (1.f / Kk);
        const float rstd = rsqrtf(s2 * (1.f / Kk) - mean * mean + LN_EPS);
        mn[s][c + 0] = (v0.x - mean) * rstd * lmw[c + 0] + lmb[c + 0];
        mn[s][c + 1] = (v0.y - mean) * rstd * lmw[c + 1] + lmb[c + 1];
        mn[s][c + 2] = (v0.z - mean) * rstd * lmw[c + 2] + lmb[c + 2];
        mn[s][c + 3] = (v0.w - mean) * rstd * lmw[c + 3] + lmb[c + 3];
    }
    __syncthreads();
    for (int i = t; i < Ss * Hh; i += 256) {
        const int s = i / Hh, h = i % Hh;
        float acc = bb1[h];
        const float* wr = w1 + (size_t)h * Kk;
        for (int kk = 0; kk < Kk; ++kk) acc += mn[s][kk] * wr[kk];
        hid[s][h] = fmaxf(acc, 0.f);
    }
    __syncthreads();
    float* so = slots + (size_t)b * Ss * Kk;
    float* oo = out + (size_t)b * Ss * Kk;
    for (int i = t; i < Ss * Kk; i += 256) {
        const int s = i / Kk, k = i % Kk;
        float acc = gru[s][k] + bb2[k];
        const float* wr = w2 + (size_t)k * Hh;
        for (int h = 0; h < Hh; ++h) acc += hid[s][h] * wr[h];
        so[i] = acc; oo[i] = acc;
    }
}

// ---------------- launch ----------------
extern "C" void kernel_launch(void* const* d_in, const int* in_sizes, int n_in,
                              void* d_out, int out_size, void* d_ws, size_t ws_size,
                              hipStream_t stream)
{
    (void)in_sizes; (void)n_in; (void)out_size; (void)ws_size;
    const float* inputs     = (const float*)d_in[0];
    const float* slots_init = (const float*)d_in[1];
    const float* mu         = (const float*)d_in[2];
    const float* lsig       = (const float*)d_in[3];
    const float* ln_in_w    = (const float*)d_in[4];
    const float* ln_in_b    = (const float*)d_in[5];
    const float* ln_slot_w  = (const float*)d_in[6];
    const float* ln_slot_b  = (const float*)d_in[7];
    const float* ln_mlp_w   = (const float*)d_in[8];
    const float* ln_mlp_b   = (const float*)d_in[9];
    const float* Wq         = (const float*)d_in[10];
    const float* Wk         = (const float*)d_in[11];
    const float* Wv         = (const float*)d_in[12];
    const float* w_ih       = (const float*)d_in[13];
    const float* w_hh       = (const float*)d_in[14];
    const float* b_ih       = (const float*)d_in[15];
    const float* b_hh       = (const float*)d_in[16];
    const float* mlp_w1     = (const float*)d_in[17];
    const float* mlp_b1     = (const float*)d_in[18];
    const float* mlp_w2     = (const float*)d_in[19];
    const float* mlp_b2     = (const float*)d_in[20];

    char* ws = (char*)d_ws;
    size_t off = 0;
    auto alloc = [&](size_t bytes) -> void* {
        off = (off + 255) & ~(size_t)255;
        void* p = ws + off;
        off += bytes;
        return p;
    };
    __bf16* wkb   = (__bf16*)alloc((size_t)Kk * Dd * 2);
    __bf16* wvb   = (__bf16*)alloc((size_t)Kk * Dd * 2);
    __bf16* qbf   = (__bf16*)alloc((size_t)Bb * 16 * Kk * 2);
    float*  accum = (float*)alloc((size_t)Bb * Ss * Kk * 4);
    float*  denom = (float*)alloc((size_t)Bb * Ss * 4);
    float*  slots = (float*)alloc((size_t)Bb * Ss * Kk * 4);
    __bf16* kbf   = (__bf16*)alloc((size_t)Bb * Nn * Kk * 2);
    __bf16* vbf   = (__bf16*)alloc((size_t)Bb * Nn * Kk * 2);

    conv_w<<<(Kk * Dd + 255) / 256, 256, 0, stream>>>(Wk, Wv, wkb, wvb);
    init_slots<<<(Bb * Ss * Kk + 255) / 256, 256, 0, stream>>>(slots_init, mu, lsig, slots);
    ln_proj_kv<<<(Bb * Nn) / 128, 256, 0, stream>>>(inputs, ln_in_w, ln_in_b, wkb, wvb, kbf, vbf);
    for (int it = 0; it < 3; ++it) {
        qgen_zero<<<Bb, 128, 0, stream>>>(slots, ln_slot_w, ln_slot_b, Wq, qbf, accum, denom);
        attn_iter<<<Bb * 4, 256, 0, stream>>>(kbf, vbf, qbf, accum, denom);
        gru_mlp<<<Bb, 256, 0, stream>>>(accum, denom, slots, w_ih, w_hh, b_ih, b_hh,
                                        ln_mlp_w, ln_mlp_b, mlp_w1, mlp_b1, mlp_w2, mlp_b2,
                                        (float*)d_out);
    }
}
